// TransformerAgent_19318762897793
// MI455X (gfx1250) — matrix operand order repaired and checked by reading, not compiled, not run on hardware
//
#include <hip/hip_runtime.h>

// Problem constants (match reference)
#define Bc   64
#define Sc   512
#define OBSc 128
#define Dc   512
#define Hc   8
#define Fc   2048
#define Lc   4
#define Ac   16
#define SR   513                // S+1 sequence rows (h0 + S embeddings)
#define MX   (Bc * SR)          // 32832 activation rows
#define HD   64                 // head dim = D/H

#define BM   64
#define BN   128
#define BK   32

typedef __attribute__((ext_vector_type(16))) __bf16 v16bf;
typedef __attribute__((ext_vector_type(8)))  float  v8f;

union FragU { v16bf v; unsigned int u[8]; };

__device__ __forceinline__ unsigned short f2bf(float f) {
  unsigned int u = __float_as_uint(f);
  u += 0x7FFFu + ((u >> 16) & 1u);        // round-to-nearest-even
  return (unsigned short)(u >> 16);
}
__device__ __forceinline__ unsigned int pack2(float a, float b) {
  return (unsigned int)f2bf(a) | ((unsigned int)f2bf(b) << 16);
}

// ---------------------------------------------------------------------------
// Generic bf16-WMMA GEMM:  C[m][n] = act( sum_k A[m,k]*B[k,n] + bias[n] )
// A addressed as A[m*sAm + k*sAk], B as B[k*sBk + n*sBn]  (arbitrary strides)
// Batch p: ptr += (p/div)*bs1 + (p%div)*bs2  for A, B, C independently.
// Output row remap: rr = (m/rdiv)*rout + m%rdiv + roff  (identity if rdiv=M).
// act: 0 none, 1 relu, 2 relu + 1e-3 (FAVOR feature map)
// Block tile 64x128, 4 waves; each wave 32x64 = 2x4 WMMA frags (8 wmma/K-step)
// ---------------------------------------------------------------------------
struct GemmP {
  const float* A; const float* B; float* C; const float* bias;
  int M, N, K;
  long long sAm, sAk; int divA; long long bsA1, bsA2;
  long long sBk, sBn; int divB; long long bsB1, bsB2;
  long long sCm;      int divC; long long bsC1, bsC2;
  int rdiv, rout, roff;
  int act;
};

__global__ __launch_bounds__(128) void gemm_bf16_wmma(GemmP p) {
  __shared__ unsigned short As[BM][BK];   // A tile, row-major [m][k]
  __shared__ unsigned short Bt[BN][BK];   // B tile, transposed [n][k]

  const int tid  = threadIdx.x;
  const int wave = tid >> 5;
  const int lane = tid & 31;
  const int half = lane >> 4;
  const int l15  = lane & 15;

  const int batch = blockIdx.z;
  const float* A = p.A + (long long)(batch / p.divA) * p.bsA1 + (long long)(batch % p.divA) * p.bsA2;
  const float* B = p.B + (long long)(batch / p.divB) * p.bsB1 + (long long)(batch % p.divB) * p.bsB2;
  float*       C = p.C + (long long)(batch / p.divC) * p.bsC1 + (long long)(batch % p.divC) * p.bsC2;

  const int m0 = blockIdx.y * BM;
  const int n0 = blockIdx.x * BN;
  const int wm = (wave >> 1) * 32;      // wave's 32x64 sub-tile
  const int wn = (wave & 1) * 64;

  const v8f vzero = {0.f,0.f,0.f,0.f,0.f,0.f,0.f,0.f};
  v8f acc[2][4];
  #pragma unroll
  for (int i = 0; i < 2; ++i)
    #pragma unroll
    for (int j = 0; j < 4; ++j) acc[i][j] = vzero;

  for (int k0 = 0; k0 < p.K; k0 += BK) {
    // prefetch next A k-slab (global_prefetch_b8)
    if (k0 + BK < p.K) {
      int m = tid >> 5, k = tid & 31;
      int gm = m0 + m, gk = k0 + BK + k;
      if (gm < p.M && gk < p.K)
        __builtin_prefetch(&A[(long long)gm * p.sAm + (long long)gk * p.sAk], 0, 3);
    }

    // ---- stage A (fp32 -> bf16) ----
    const bool fastA = (p.sAk == 1) && (m0 + BM <= p.M) && (k0 + BK <= p.K);
    if (fastA) {
      // 64x32 floats = 512 float4; 4 per thread: b128 load + b64 LDS store
      #pragma unroll
      for (int it = 0; it < 4; ++it) {
        int fid = tid + it * 128;
        int row = fid >> 3, quad = fid & 7;
        const float4 f = *(const float4*)(A + (long long)(m0 + row) * p.sAm + (k0 + quad * 4));
        unsigned long long dd = (unsigned long long)pack2(f.x, f.y) |
                                ((unsigned long long)pack2(f.z, f.w) << 32);
        *(unsigned long long*)&As[row][quad * 4] = dd;
      }
    } else {
      for (int i = tid; i < BM * BK; i += 128) {
        int m = i >> 5, k = i & 31;
        int gm = m0 + m, gk = k0 + k;
        float v = 0.f;
        if (gm < p.M && gk < p.K) v = A[(long long)gm * p.sAm + (long long)gk * p.sAk];
        As[m][k] = f2bf(v);
      }
    }

    // ---- stage B transposed ----
    const bool fastB = (p.sBn == 1) && (n0 + BN <= p.N) && (k0 + BK <= p.K);
    if (fastB) {
      // 32x128 floats = 1024 float4; 8 per thread: b128 load, 4 b16 LDS stores
      #pragma unroll
      for (int it = 0; it < 8; ++it) {
        int fid = tid + it * 128;
        int k = fid >> 5, nq = fid & 31;
        const float4 f = *(const float4*)(B + (long long)(k0 + k) * p.sBk + (n0 + nq * 4));
        int nb = nq * 4;
        Bt[nb + 0][k] = f2bf(f.x);
        Bt[nb + 1][k] = f2bf(f.y);
        Bt[nb + 2][k] = f2bf(f.z);
        Bt[nb + 3][k] = f2bf(f.w);
      }
    } else {
      for (int i = tid; i < BN * BK; i += 128) {
        int n = i >> 5, k = i & 31;
        int gn = n0 + n, gk = k0 + k;
        float v = 0.f;
        if (gn < p.N && gk < p.K) v = B[(long long)gk * p.sBk + (long long)gn * p.sBn];
        Bt[n][k] = f2bf(v);
      }
    }
    __syncthreads();

    // Build fragments per ISA 7.12.2 (16-bit 16x32 A layout; B K-pairs per lane)
    FragU a[2], b[4];
    #pragma unroll
    for (int s = 0; s < 2; ++s) {
      int row = wm + s * 16 + l15;
      #pragma unroll
      for (int j = 0; j < 8; ++j) {
        int ka = 2 * (j & 3) + 8 * half + 16 * (j >> 2);   // A K swizzle
        a[s].u[j] = *(const unsigned int*)&As[row][ka];
      }
    }
    #pragma unroll
    for (int t = 0; t < 4; ++t) {
      int col = wn + t * 16 + l15;
      #pragma unroll
      for (int j = 0; j < 8; ++j) {
        int kb = 2 * (j & 3) + 8 * half + 16 * (j >> 2);   // B K pairs
        b[t].u[j] = *(const unsigned int*)&Bt[col][kb];
      }
    }
    #pragma unroll
    for (int i = 0; i < 2; ++i)
      #pragma unroll
      for (int j = 0; j < 4; ++j)
        acc[i][j] = __builtin_amdgcn_wmma_f32_16x16x32_bf16(
            false, a[i].v, false, b[j].v, (short)0, acc[i][j], false, false);
    __syncthreads();
  }

  // Epilogue: bias + activation + guarded, remapped store
  #pragma unroll
  for (int i = 0; i < 2; ++i) {
    #pragma unroll
    for (int j = 0; j < 4; ++j) {
      int n = n0 + wn + j * 16 + l15;
      if (n >= p.N) continue;
      float bv = p.bias ? p.bias[n] : 0.f;
      #pragma unroll
      for (int r = 0; r < 8; ++r) {
        int m = m0 + wm + i * 16 + r + 8 * half;   // C layout: VGPR r -> M=r(+8)
        if (m >= p.M) continue;
        float v = acc[i][j][r] + bv;
        if (p.act == 1)      v = fmaxf(v, 0.f);
        else if (p.act == 2) v = fmaxf(v, 0.f) + 0.001f;
        long long rr = (long long)(m / p.rdiv) * p.rout + (m % p.rdiv) + p.roff;
        C[rr * p.sCm + n] = v;
      }
    }
  }
}

// ---------------------------------------------------------------------------
// Small fp32 helper kernels
// ---------------------------------------------------------------------------

// x row 0 per batch: h0 = resets ? 0 : hidden_state
__global__ void k_init_x(const float* __restrict__ hidden, const int* __restrict__ resets,
                         float* __restrict__ x) {
  int idx = blockIdx.x * blockDim.x + threadIdx.x;
  if (idx >= Bc * Dc) return;
  int b = idx >> 9;
  int d = idx & (Dc - 1);
  float v = resets[b] ? 0.f : hidden[idx];
  x[(long long)b * SR * Dc + d] = v;
}

// ksum[b][d] = sum_s pk[b][s][d]   (d = h*64+m)
__global__ void k_ksum(const float* __restrict__ pk, float* __restrict__ ksum) {
  int idx = blockIdx.x * blockDim.x + threadIdx.x;
  if (idx >= Bc * Dc) return;
  int b = idx >> 9;
  int d = idx & (Dc - 1);
  const float* base = pk + (long long)b * SR * Dc + d;
  float s = 0.f;
  for (int t = 0; t < SR; ++t) s += base[(long long)t * Dc];
  ksum[idx] = s;
}

// per (b,s,h): den = dot64(pq, ksum); attout = num / den  (attout may alias pq)
__global__ __launch_bounds__(64) void k_den_div(const float* __restrict__ pq,
                                                const float* __restrict__ num,
                                                const float* __restrict__ ksum,
                                                float* __restrict__ attout) {
  long long blk = blockIdx.x;           // b*SR*H + s*H + h
  int h = (int)(blk % Hc);
  long long bs = blk / Hc;              // b*SR + s
  int b = (int)(bs / SR);
  int t = threadIdx.x;                  // 0..63
  long long rowoff = bs * Dc + h * HD;
  float q  = pq[rowoff + t];
  float ks = ksum[(long long)b * Dc + h * HD + t];
  __shared__ float red[64];
  red[t] = q * ks;
  __syncthreads();
  for (int o = 32; o > 0; o >>= 1) { if (t < o) red[t] += red[t + o]; __syncthreads(); }
  float den = red[0];
  float nv = num[rowoff + t];
  attout[rowoff + t] = nv / den;
}

// x = LayerNorm(y + x) * s + b   (one 256-thread block per row, D=512)
__global__ __launch_bounds__(256) void k_ln(const float* __restrict__ y, float* __restrict__ x,
                                            const float* __restrict__ sc, const float* __restrict__ sh) {
  int row = blockIdx.x;
  int t = threadIdx.x;
  long long off = (long long)row * Dc;
  float v0 = y[off + t]       + x[off + t];
  float v1 = y[off + t + 256] + x[off + t + 256];
  __shared__ float red[256];
  red[t] = v0 + v1;
  __syncthreads();
  for (int o = 128; o > 0; o >>= 1) { if (t < o) red[t] += red[t + o]; __syncthreads(); }
  float mean = red[0] * (1.f / Dc);
  __syncthreads();
  float d0 = v0 - mean, d1 = v1 - mean;
  red[t] = d0 * d0 + d1 * d1;
  __syncthreads();
  for (int o = 128; o > 0; o >>= 1) { if (t < o) red[t] += red[t + o]; __syncthreads(); }
  float inv = rsqrtf(red[0] * (1.f / Dc) + 1e-6f);
  x[off + t]       = d0 * inv * sc[t]       + sh[t];
  x[off + t + 256] = d1 * inv * sc[t + 256] + sh[t + 256];
}

// h_out = x[:,0,:] ; q_vals = h_out @ W_qp + b_qp ; packed [B*D | B*A]
__global__ __launch_bounds__(256) void k_head(const float* __restrict__ x,
                                              const float* __restrict__ Wqp,
                                              const float* __restrict__ bqp,
                                              float* __restrict__ out) {
  int b = blockIdx.x;
  __shared__ float row[Dc];
  const float* xr = x + (long long)b * SR * Dc;
  for (int i = threadIdx.x; i < Dc; i += 256) {
    float v = xr[i];
    row[i] = v;
    out[b * Dc + i] = v;
  }
  __syncthreads();
  if (threadIdx.x < Ac) {
    int a = threadIdx.x;
    float s = bqp[a];
    for (int d = 0; d < Dc; ++d) s += row[d] * Wqp[d * Ac + a];
    out[Bc * Dc + b * Ac + a] = s;
  }
}

// ---------------------------------------------------------------------------
// Host orchestration
// ---------------------------------------------------------------------------
static GemmP simpleP(const float* A, const float* B, float* C, const float* bias,
                     int M, int N, int K, int lda, int ldb, int ldc, int act) {
  GemmP p{};
  p.A = A; p.B = B; p.C = C; p.bias = bias;
  p.M = M; p.N = N; p.K = K;
  p.sAm = lda; p.sAk = 1; p.divA = 1; p.bsA1 = 0; p.bsA2 = 0;
  p.sBk = ldb; p.sBn = 1; p.divB = 1; p.bsB1 = 0; p.bsB2 = 0;
  p.sCm = ldc;             p.divC = 1; p.bsC1 = 0; p.bsC2 = 0;
  p.rdiv = M; p.rout = M; p.roff = 0;
  p.act = act;
  return p;
}

extern "C" void kernel_launch(void* const* d_in, const int* in_sizes, int n_in,
                              void* d_out, int out_size, void* d_ws, size_t ws_size,
                              hipStream_t stream) {
  (void)in_sizes; (void)n_in; (void)out_size; (void)ws_size;
  const float* hidden = (const float*)d_in[0];
  const float* ins    = (const float*)d_in[1];
  const int*   resets = (const int*)d_in[2];
  const float* W_emb  = (const float*)d_in[3];
  const float* b_emb  = (const float*)d_in[4];
  const float* Wq     = (const float*)d_in[5];
  const float* bq     = (const float*)d_in[6];
  const float* Wk     = (const float*)d_in[7];
  const float* bk     = (const float*)d_in[8];
  const float* Wv     = (const float*)d_in[9];
  const float* bv     = (const float*)d_in[10];
  const float* Wo     = (const float*)d_in[11];
  const float* bo     = (const float*)d_in[12];
  const float* ln1_s  = (const float*)d_in[13];
  const float* ln1_b  = (const float*)d_in[14];
  const float* ln2_s  = (const float*)d_in[15];
  const float* ln2_b  = (const float*)d_in[16];
  const float* W1     = (const float*)d_in[17];
  const float* b1     = (const float*)d_in[18];
  const float* W2     = (const float*)d_in[19];
  const float* b2     = (const float*)d_in[20];
  const float* W_qp   = (const float*)d_in[21];
  const float* b_qp   = (const float*)d_in[22];
  float* out = (float*)d_out;

  // Workspace layout (floats)
  float* ws   = (float*)d_ws;
  const size_t ACT = (size_t)MX * Dc;            // 16,809,984 floats
  float* x    = ws;                              // sequence activations (B,513,D)
  float* qb   = ws + 1 * ACT;                    // pq; later aliased as attout
  float* kb   = ws + 2 * ACT;                    // pk; later Wo/FFN2 output
  float* vb   = ws + 3 * ACT;                    // v
  float* nb   = ws + 4 * ACT;                    // num; later FFN hidden chunks
  float* kv   = ws + 5 * ACT;                    // (B*H, 64, 64)
  float* ksum = kv + (size_t)Bc * Hc * HD * HD;  // (B, D)

  const long long rowsBH = (long long)SR * Dc;   // per-batch activation stride

  // 1) x row 0 = h0 (with resets)
  k_init_x<<<dim3((Bc * Dc + 255) / 256), dim3(256), 0, stream>>>(hidden, resets, x);

  // 2) embedding GEMM -> x rows 1..512 per batch (row remap 512 -> 513, +1)
  {
    GemmP p = simpleP(ins, W_emb, x, b_emb, Bc * Sc, Dc, OBSc, OBSc, Dc, Dc, 0);
    p.rdiv = Sc; p.rout = SR; p.roff = 1;
    gemm_bf16_wmma<<<dim3(Dc / BN, (Bc * Sc) / BM, 1), dim3(128), 0, stream>>>(p);
  }

  const dim3 gBig(Dc / BN, (MX + BM - 1) / BM, 1);   // (4, 513)
  for (int l = 0; l < Lc; ++l) {
    const float* Wql = Wq + (size_t)l * Dc * Dc;
    const float* Wkl = Wk + (size_t)l * Dc * Dc;
    const float* Wvl = Wv + (size_t)l * Dc * Dc;
    const float* Wol = Wo + (size_t)l * Dc * Dc;
    const float* W1l = W1 + (size_t)l * Dc * Fc;
    const float* W2l = W2 + (size_t)l * Fc * Dc;

    // 3a) pq = relu(x@Wq + bq) + eps ; pk likewise ; v plain
    gemm_bf16_wmma<<<gBig, dim3(128), 0, stream>>>(
        simpleP(x, Wql, qb, bq + l * Dc, MX, Dc, Dc, Dc, Dc, Dc, 2));
    gemm_bf16_wmma<<<gBig, dim3(128), 0, stream>>>(
        simpleP(x, Wkl, kb, bk + l * Dc, MX, Dc, Dc, Dc, Dc, Dc, 2));
    gemm_bf16_wmma<<<gBig, dim3(128), 0, stream>>>(
        simpleP(x, Wvl, vb, bv + l * Dc, MX, Dc, Dc, Dc, Dc, Dc, 0));

    // 3b) kv[p=(b,h)] = pk^T @ v : M=64, N=64, K=513, batched over 512
    {
      GemmP p{};
      p.A = kb; p.B = vb; p.C = kv; p.bias = nullptr;
      p.M = HD; p.N = HD; p.K = SR;
      p.sAm = 1;  p.sAk = Dc; p.divA = Hc; p.bsA1 = rowsBH; p.bsA2 = HD;
      p.sBk = Dc; p.sBn = 1;  p.divB = Hc; p.bsB1 = rowsBH; p.bsB2 = HD;
      p.sCm = HD;             p.divC = 1;  p.bsC1 = HD * HD; p.bsC2 = 0;
      p.rdiv = HD; p.rout = HD; p.roff = 0; p.act = 0;
      gemm_bf16_wmma<<<dim3(1, 1, Bc * Hc), dim3(128), 0, stream>>>(p);
    }

    // 3c) ksum
    k_ksum<<<dim3((Bc * Dc + 255) / 256), dim3(256), 0, stream>>>(kb, ksum);

    // 3d) num[p=(b,h)] = pq @ kv : M=513, N=64, K=64, batched over 512
    {
      GemmP p{};
      p.A = qb; p.B = kv; p.C = nb; p.bias = nullptr;
      p.M = SR; p.N = HD; p.K = HD;
      p.sAm = Dc; p.sAk = 1; p.divA = Hc; p.bsA1 = rowsBH; p.bsA2 = HD;
      p.sBk = HD; p.sBn = 1; p.divB = 1;  p.bsB1 = HD * HD; p.bsB2 = 0;
      p.sCm = Dc;            p.divC = Hc; p.bsC1 = rowsBH; p.bsC2 = HD;
      p.rdiv = SR; p.rout = SR; p.roff = 0; p.act = 0;
      gemm_bf16_wmma<<<dim3(1, (SR + BM - 1) / BM, Bc * Hc), dim3(128), 0, stream>>>(p);
    }

    // 3e) den + divide (attout written in place over pq)
    k_den_div<<<dim3(Bc * SR * Hc), dim3(64), 0, stream>>>(qb, nb, ksum, qb);

    // 3f) Wo projection -> kb
    gemm_bf16_wmma<<<gBig, dim3(128), 0, stream>>>(
        simpleP(qb, Wol, kb, bo + l * Dc, MX, Dc, Dc, Dc, Dc, Dc, 0));

    // 3g) x = LN(att + x)
    k_ln<<<dim3(MX), dim3(256), 0, stream>>>(kb, x, ln1_s + l * Dc, ln1_b + l * Dc);

    // 3h) FFN in 4 row-chunks (hidden reuses nb): h1 = relu(x@W1+b1); y = h1@W2+b2 -> kb
    const int CH = MX / 4;                        // 8208 rows/chunk
    for (int c = 0; c < 4; ++c) {
      const float* xa = x  + (size_t)c * CH * Dc;
      float*       ya = kb + (size_t)c * CH * Dc;
      gemm_bf16_wmma<<<dim3(Fc / BN, (CH + BM - 1) / BM, 1), dim3(128), 0, stream>>>(
          simpleP(xa, W1l, nb, b1 + l * Fc, CH, Fc, Dc, Dc, Fc, Fc, 1));
      gemm_bf16_wmma<<<dim3(Dc / BN, (CH + BM - 1) / BM, 1), dim3(128), 0, stream>>>(
          simpleP(nb, W2l, ya, b2 + l * Dc, CH, Dc, Fc, Fc, Dc, Dc, 0));
    }

    // 3i) x = LN(ff + x)
    k_ln<<<dim3(MX), dim3(256), 0, stream>>>(kb, x, ln2_s + l * Dc, ln2_b + l * Dc);
  }

  // 4) head: h_out and q_vals
  k_head<<<dim3(Bc), dim3(256), 0, stream>>>(x, W_qp, b_qp, out);
}
